// LearnedUpdate_3092376453783
// MI455X (gfx1250) — matrix operand order repaired and checked by reading, not compiled, not run on hardware
//
#include <hip/hip_runtime.h>
#include <math.h>

// ---------------------------------------------------------------------------
// LearnedUpdate fused kernel for MI455X (gfx1250, wave32, WMMA)
//
// Roofline: 67 MB x/grad read + 33.5 MB out write @ 23.3 TB/s => ~4.4us floor.
// GEMM1 (4096 x 4097 x 80, 2.7 GFLOP) runs on v_wmma_f32_16x16x32_bf16 with
// raw (unnormalized) bf16 data; per-sample inf-norms are folded in at the
// epilogue (normalization is linear), and are computed inside the K-loop so
// x/grad are read exactly once. Epilogue transcendentals use the hardware
// v_exp_f32 path to keep the per-tile serial tail short.
// ---------------------------------------------------------------------------

typedef __attribute__((ext_vector_type(16))) __bf16 v16bf;
typedef __attribute__((ext_vector_type(8)))  float  v8f;
typedef __attribute__((ext_vector_type(4)))  float  f4;

#define D_DIM 2048
#define H_DIM 20
#define NG    80          // 4*H gates
#define KC    4096        // combined K (x part + grad part)
#define KPW   512         // K per wave (8 waves)
#define RHO_F 0.99f

// workspace (float view):
//   [0:80)    bias1 = b_ih1 + b_hh1 + W_hh1 @ h1_0
//   [80:160)  bias2 = b_ih2 + b_hh2 + W_hh2 @ h2_0
//   [160:240) Wl    = W_ih1[:, 2048]  (loss column)
//   [240]     scale = softplus(alpha).poly(t) * RHO^t
//   bf16 packed weights [80][4096] at float offset 256

// fast transcendentals on the v_exp_f32 path; saturate correctly at +-inf
__device__ __forceinline__ float sigf(float x)  { return 1.0f / (1.0f + __expf(-x)); }
__device__ __forceinline__ float ftanh(float x) { return 1.0f - 2.0f / (__expf(2.0f * x) + 1.0f); }
__device__ __forceinline__ float amax4(f4 v) {
    return fmaxf(fmaxf(fabsf(v.x), fabsf(v.y)), fmaxf(fabsf(v.z), fabsf(v.w)));
}

// ------------------------- prep kernel (tiny) ------------------------------
__global__ __launch_bounds__(256)
void lu_prep_kernel(const float* __restrict__ W_ih1, const float* __restrict__ b_ih1,
                    const float* __restrict__ W_hh1, const float* __restrict__ b_hh1,
                    const float* __restrict__ b_ih2,
                    const float* __restrict__ W_hh2, const float* __restrict__ b_hh2,
                    const float* __restrict__ h1_0,  const float* __restrict__ h2_0,
                    const float* __restrict__ alpha_raw, const int* __restrict__ t_ptr,
                    float* __restrict__ ws_f, __bf16* __restrict__ wpack)
{
    const int tid = threadIdx.x;
    const int ROWLEN = 2 * D_DIM + 1;   // 4097

    // pack [Wx | Wg] row-major bf16: combined k<2048 -> col k, else col k+1
    for (int idx = tid; idx < NG * KC; idx += 256) {
        int g = idx >> 12;
        int k = idx & (KC - 1);
        int col = (k < D_DIM) ? k : (k + 1);
        wpack[idx] = (__bf16)W_ih1[(size_t)g * ROWLEN + col];
    }

    if (tid < NG) {
        float s1 = b_ih1[tid] + b_hh1[tid];
        float s2 = b_ih2[tid] + b_hh2[tid];
        #pragma unroll
        for (int h = 0; h < H_DIM; ++h) {
            s1 = fmaf(W_hh1[tid * H_DIM + h], h1_0[h], s1);
            s2 = fmaf(W_hh2[tid * H_DIM + h], h2_0[h], s2);
        }
        ws_f[tid]       = s1;
        ws_f[80 + tid]  = s2;
        ws_f[160 + tid] = W_ih1[(size_t)tid * ROWLEN + D_DIM];   // loss column
    }

    if (tid == 0) {
        float tf = (float)t_ptr[0];
        float poly = 0.0f, tp = 1.0f;
        #pragma unroll
        for (int j = 0; j < 3; ++j) {
            float a  = alpha_raw[j];
            float sp = (a > 20.0f) ? a : log1pf(__expf(a));  // softplus
            poly = fmaf(sp, tp, poly);
            tp *= tf;
        }
        ws_f[240] = poly * __powf(RHO_F, tf);
    }
}

// --------------------- fused GEMM + LSTM + output --------------------------
__global__ __launch_bounds__(256)
void lu_main_kernel(const float* __restrict__ x,     const float* __restrict__ loss,
                    const float* __restrict__ grad,  const __bf16* __restrict__ wpack,
                    const float* __restrict__ ws_f,  const float* __restrict__ W_ih2,
                    const float* __restrict__ c1_0,  const float* __restrict__ c2_0,
                    const float* __restrict__ W_out, const float* __restrict__ b_out,
                    float* __restrict__ out)
{
    __shared__ float red_c[8 * 5 * 8 * 32];   // per-wave C fragments (40 KB)
    __shared__ float red_mx[8 * 32];          // per-wave per-lane |.|max partials
    __shared__ float g_x[16 * NG];            // reduced x-part gate pre-acts
    __shared__ float g_g[16 * NG];            // reduced grad-part gate pre-acts
    __shared__ float h2s[16 * H_DIM];
    __shared__ float scl[16];

    const int tid  = threadIdx.x;
    const int w    = tid >> 5;        // wave id 0..7
    const int lid  = tid & 31;
    const int mrow = lid & 15;        // M row within tile / N col within tile
    const int half = lid >> 4;        // lane half per ISA 16-bit fragment layout
    const int m0   = blockIdx.x * 16;

    // each wave owns a 512-wide slice of combined K; waves 0-3 -> x, 4-7 -> grad
    const bool  isX   = (w < 4);
    const float* src  = isX ? x : grad;
    const int klocal0 = (isX ? w : (w - 4)) * KPW;
    const int kcomb0  = w * KPW;
    const float* arow = src + (size_t)(m0 + mrow) * D_DIM + klocal0;

    v8f acc[5];
    #pragma unroll
    for (int nt = 0; nt < 5; ++nt) { v8f z = {}; acc[nt] = z; }
    float amax = 0.0f;

    for (int kb = 0; kb < KPW; kb += 32) {
        __builtin_prefetch(arow + kb + 64, 0, 1);   // prefetch ahead on A stream

        // A fragment: lane holds 16 contiguous K at (kb + half*16),
        // matching the B fragment order
        f4 a0 = *(const f4*)(arow + kb + half * 16);
        f4 a1 = *(const f4*)(arow + kb + half * 16 + 4);
        f4 a2 = *(const f4*)(arow + kb + half * 16 + 8);
        f4 a3 = *(const f4*)(arow + kb + half * 16 + 12);

        amax = fmaxf(amax, fmaxf(fmaxf(amax4(a0), amax4(a1)),
                                 fmaxf(amax4(a2), amax4(a3))));

        v16bf afrag;
        afrag[0]  = (__bf16)a0.x; afrag[1]  = (__bf16)a0.y;
        afrag[2]  = (__bf16)a0.z; afrag[3]  = (__bf16)a0.w;
        afrag[4]  = (__bf16)a1.x; afrag[5]  = (__bf16)a1.y;
        afrag[6]  = (__bf16)a1.z; afrag[7]  = (__bf16)a1.w;
        afrag[8]  = (__bf16)a2.x; afrag[9]  = (__bf16)a2.y;
        afrag[10] = (__bf16)a2.z; afrag[11] = (__bf16)a2.w;
        afrag[12] = (__bf16)a3.x; afrag[13] = (__bf16)a3.y;
        afrag[14] = (__bf16)a3.z; afrag[15] = (__bf16)a3.w;

        const int kc = kcomb0 + kb;
        #pragma unroll
        for (int nt = 0; nt < 5; ++nt) {
            // B fragment: lane = gate column (nt*16 + mrow), 16 contiguous K
            const v16bf bfrag =
                *(const v16bf*)(wpack + (size_t)(nt * 16 + mrow) * KC + kc + half * 16);
            acc[nt] = __builtin_amdgcn_wmma_f32_16x16x32_bf16(
                false, afrag, false, bfrag, (short)0, acc[nt], false, false);
        }
    }

    // dump per-wave partial C + maxabs to LDS
    #pragma unroll
    for (int nt = 0; nt < 5; ++nt)
        #pragma unroll
        for (int v = 0; v < 8; ++v)
            red_c[((w * 5 + nt) * 8 + v) * 32 + lid] = acc[nt][v];
    red_mx[w * 32 + lid] = amax;
    __syncthreads();

    // reduce 8 waves -> 16x80 gate pre-activations (x part and grad part)
    for (int i = tid; i < 16 * NG; i += 256) {
        int r = i / NG, n = i - r * NG;
        int nt = n >> 4, nl = n & 15;
        int v = r & 7, lane = nl + ((r >> 3) << 4);   // C fragment layout
        float sx = 0.0f, sg = 0.0f;
        #pragma unroll
        for (int ww = 0; ww < 4; ++ww) sx += red_c[((ww * 5 + nt) * 8 + v) * 32 + lane];
        #pragma unroll
        for (int ww = 4; ww < 8; ++ww) sg += red_c[((ww * 5 + nt) * 8 + v) * 32 + lane];
        g_x[i] = sx; g_g[i] = sg;
    }
    __syncthreads();

    // both LSTM cells (H=20) for the 16 rows, one thread per row
    if (tid < 16) {
        const int r = tid;
        float ax = 0.0f, ag = 0.0f;
        #pragma unroll
        for (int ww = 0; ww < 4; ++ww)
            ax = fmaxf(ax, fmaxf(red_mx[ww * 32 + r], red_mx[ww * 32 + r + 16]));
        #pragma unroll
        for (int ww = 4; ww < 8; ++ww)
            ag = fmaxf(ag, fmaxf(red_mx[ww * 32 + r], red_mx[ww * 32 + r + 16]));
        if (!(ax > 0.0f)) ax = 1.0f;
        if (!(ag > 0.0f)) ag = 1.0f;
        const float rax = 1.0f / ax, rag = 1.0f / ag;

        const float lo = loss[m0 + r];
        const float al = fabsf(lo);
        const float ls = (al > 0.0f) ? lo / al : 0.0f;   // loss / max|loss|

        float h1v[H_DIM];
        #pragma unroll
        for (int h = 0; h < H_DIM; ++h) {
            float gi = g_x[r*NG +      h] * rax + g_g[r*NG +      h] * rag + ls * ws_f[160 +      h] + ws_f[     h];
            float gf = g_x[r*NG + 20 + h] * rax + g_g[r*NG + 20 + h] * rag + ls * ws_f[160 + 20 + h] + ws_f[20 + h];
            float gg = g_x[r*NG + 40 + h] * rax + g_g[r*NG + 40 + h] * rag + ls * ws_f[160 + 40 + h] + ws_f[40 + h];
            float go = g_x[r*NG + 60 + h] * rax + g_g[r*NG + 60 + h] * rag + ls * ws_f[160 + 60 + h] + ws_f[60 + h];
            float c1n = sigf(gf) * c1_0[h] + sigf(gi) * ftanh(gg);
            h1v[h] = sigf(go) * ftanh(c1n);
        }
        #pragma unroll
        for (int h = 0; h < H_DIM; ++h) {
            float s0 = ws_f[80 +      h], s1 = ws_f[80 + 20 + h];
            float s2 = ws_f[80 + 40 + h], s3 = ws_f[80 + 60 + h];
            #pragma unroll
            for (int hh = 0; hh < H_DIM; ++hh) {
                float hv = h1v[hh];
                s0 = fmaf(W_ih2[(     h) * H_DIM + hh], hv, s0);
                s1 = fmaf(W_ih2[(20 + h) * H_DIM + hh], hv, s1);
                s2 = fmaf(W_ih2[(40 + h) * H_DIM + hh], hv, s2);
                s3 = fmaf(W_ih2[(60 + h) * H_DIM + hh], hv, s3);
            }
            float c2n = sigf(s1) * c2_0[h] + sigf(s0) * ftanh(s2);
            h2s[r * H_DIM + h] = sigf(s3) * ftanh(c2n);
        }
        scl[r] = ax * ws_f[240];   // inf-norm rescale * poly * rho^t
    }
    __syncthreads();

    // output tile: out[m0+r][d] = tanh(h2 . W_out[d] + b_out[d]) * scl[r]
    for (int d = tid; d < D_DIM; d += 256) {
        float wv[H_DIM];
        #pragma unroll
        for (int h = 0; h < H_DIM; ++h) wv[h] = W_out[(size_t)d * H_DIM + h];
        const float bo = b_out[d];
        #pragma unroll
        for (int r = 0; r < 16; ++r) {
            float s = bo;
            #pragma unroll
            for (int h = 0; h < H_DIM; ++h) s = fmaf(h2s[r * H_DIM + h], wv[h], s);
            out[(size_t)(m0 + r) * D_DIM + d] = ftanh(s) * scl[r];
        }
    }
}

// --------------------------------------------------------------------------
extern "C" void kernel_launch(void* const* d_in, const int* in_sizes, int n_in,
                              void* d_out, int out_size, void* d_ws, size_t ws_size,
                              hipStream_t stream) {
    (void)in_sizes; (void)n_in; (void)out_size; (void)ws_size;
    const float* x      = (const float*)d_in[0];
    const float* loss   = (const float*)d_in[1];
    const float* grad   = (const float*)d_in[2];
    const float* W_ih1  = (const float*)d_in[3];
    const float* b_ih1  = (const float*)d_in[4];
    const float* W_hh1  = (const float*)d_in[5];
    const float* b_hh1  = (const float*)d_in[6];
    const float* W_ih2  = (const float*)d_in[7];
    const float* b_ih2  = (const float*)d_in[8];
    const float* W_hh2  = (const float*)d_in[9];
    const float* b_hh2  = (const float*)d_in[10];
    const float* W_out  = (const float*)d_in[11];
    const float* b_out  = (const float*)d_in[12];
    const float* h1_0   = (const float*)d_in[13];
    const float* c1_0   = (const float*)d_in[14];
    const float* h2_0   = (const float*)d_in[15];
    const float* c2_0   = (const float*)d_in[16];
    const float* alpha  = (const float*)d_in[17];
    const int*   t_ptr  = (const int*)d_in[18];

    float*  out   = (float*)d_out;
    float*  ws_f  = (float*)d_ws;
    __bf16* wpack = (__bf16*)(ws_f + 256);   // 80*4096 bf16 = 640 KB

    lu_prep_kernel<<<1, 256, 0, stream>>>(W_ih1, b_ih1, W_hh1, b_hh1,
                                          b_ih2, W_hh2, b_hh2,
                                          h1_0, h2_0, alpha, t_ptr,
                                          ws_f, wpack);
    lu_main_kernel<<<256, 256, 0, stream>>>(x, loss, grad, wpack, ws_f,
                                            W_ih2, c1_0, c2_0, W_out, b_out, out);
}
